// TimeNet_60103772340634
// MI455X (gfx1250) — matrix operand order repaired and checked by reading, not compiled, not run on hardware
//
#include <hip/hip_runtime.h>
#include <hip/hip_bf16.h>

// ---------------------------------------------------------------------------
// Seq2seq GRU (B=1024, T=512, F=1, H=60) for MI455X / gfx1250, wave32 WMMA.
//
// Strategy:
//  * One wave owns 16 batch rows; the GRU recurrence h@Whh^T is row-local in
//    batch, so the whole 512-step chain runs with NO cross-wave sync.
//  * Gates padded to 64 cols each -> N=192 = 12 WMMA N-tiles; r/z/n columns
//    are lane-aligned in the f32 D-layout (lane owns col c=j*16+p).
//  * bf16 WMMA f32_16x16x32; K=60 padded to 64 -> 2 WMMA per tile per GEMM.
//  * Weights pre-staged in LDS in exact B-fragment layout (loaded once).
//  * D-layout -> A-layout transpose of h via a 2KB wave-private LDS tile,
//    guarded with per-wave s_wait_dscnt (no block barriers in the T loop).
//  * Inter-layer activations in workspace as bf16 [T][B][64] (K-padded) so
//    A-fragments load directly as two 16B chunks from global memory.
// ---------------------------------------------------------------------------

#define DEV static __device__ __forceinline__

typedef __attribute__((ext_vector_type(16))) __bf16 v16bf;
typedef __attribute__((ext_vector_type(8)))  __bf16 v8bf;
typedef __attribute__((ext_vector_type(8)))  float  v8f;

static constexpr int Bsz = 1024;
static constexpr int Tsz = 512;
static constexpr int Hsz = 60;
static constexpr int HP  = 64;     // padded K
static constexpr int NT  = 12;     // N tiles (3 gates x 4)

DEV v8f v8f_zero() {
  v8f z;
#pragma unroll
  for (int i = 0; i < 8; ++i) z[i] = 0.f;
  return z;
}

// Build a 16x32 bf16 A-fragment from a row-major [16][64] bf16 tile.
// A layout (16-bit, wave32): lanes 0-15: m=lane, elems 0..7 -> K=kb+0..7,
// elems 8..15 -> K=kb+16..23 with kb=khalf*32; lanes 16-31: kb += 8.
DEV v16bf load_a(const __bf16* base, int lane, int khalf) {
  int m  = lane & 15;
  int kb = khalf * 32 + ((lane & 16) ? 8 : 0);
  v8bf lo = *(const v8bf*)(base + m * HP + kb);
  v8bf hi = *(const v8bf*)(base + m * HP + kb + 16);
  return __builtin_shufflevector(lo, hi, 0, 1, 2, 3, 4, 5, 6, 7,
                                 8, 9, 10, 11, 12, 13, 14, 15);
}

// Fill LDS B-fragments for weight W [180 x ld] (row-major, ld = 60).
// B layout (32x16 bf16): lane L -> col n=L&15, K base = khalf*32 + (L<16?0:8),
// elems 0..7 -> K=kb+0..7, elems 8..15 -> K=kb+16..23. Padded-gate mapping: tile tl -> gate g=tl/4,
// col-in-gate c = (tl&3)*16+n, original row o = g*60+c (zero if c>=60).
DEV void fill_bfrags(__bf16* dst, const float* __restrict__ W, int ld,
                     int tid, int nthr) {
  for (int slot = tid; slot < NT * 2 * 32; slot += nthr) {
    int tl = slot >> 6;
    int rem = slot & 63;
    int kh = rem >> 5;
    int lr = rem & 31;
    int n  = lr & 15;
    int g  = tl >> 2, j = tl & 3;
    int c  = j * 16 + n;
    int kb = kh * 32 + ((lr & 16) ? 8 : 0);
    int o  = g * 60 + (c < 60 ? c : 0);
#pragma unroll
    for (int e = 0; e < 16; ++e) {
      int k = kb + (e < 8 ? e : e + 8);
      float v = (c < Hsz && k < Hsz) ? W[(size_t)o * ld + k] : 0.f;
      dst[(size_t)slot * 16 + e] = (__bf16)v;
    }
  }
}

// 12-tile GEMM-accumulate: tiles 0..7 (r,z gates) -> acc[0..7];
// tiles 8..11 (n gate) -> acc[nbase..nbase+3]  (nbase=8 for hh, 12 for ih).
DEV void gemm12(v8f acc[16], int nbase, const __bf16* bfr,
                v16bf a0, v16bf a1, int lane) {
#pragma unroll
  for (int tl = 0; tl < NT; ++tl) {
    int ai = (tl < 8) ? tl : (nbase + tl - 8);
    v16bf b0 = *(const v16bf*)(bfr + ((size_t)((tl * 2 + 0) * 32 + lane)) * 16);
    v16bf b1 = *(const v16bf*)(bfr + ((size_t)((tl * 2 + 1) * 32 + lane)) * 16);
    v8f c = acc[ai];
    c = __builtin_amdgcn_wmma_f32_16x16x32_bf16(false, a0, false, b0,
                                                (short)0, c, false, false);
    c = __builtin_amdgcn_wmma_f32_16x16x32_bf16(false, a1, false, b1,
                                                (short)0, c, false, false);
    acc[ai] = c;
  }
}

DEV float sigmoidf_(float x) { return 1.f / (1.f + __expf(-x)); }

// GRU gate math in D-layout: lane owns col c=j*16+p for rows m=r(+8*half).
template <bool HASX>
DEV void gates_update(float hold[4][8], const v8f acc[16],
                      const float brz0[4], const float brz1[4],
                      const float bin[4], const float bhn[4],
                      const float wx[12], const float xv[8], int p) {
#pragma unroll
  for (int j = 0; j < 4; ++j) {
    bool valid = (j * 16 + p) < Hsz;
#pragma unroll
    for (int r = 0; r < 8; ++r) {
      float gr  = acc[j][r]      + brz0[j];
      float gz  = acc[4 + j][r]  + brz1[j];
      float gni = acc[12 + j][r] + bin[j];
      float gnh = acc[8 + j][r]  + bhn[j];
      if (HASX) {
        gr  += xv[r] * wx[j];
        gz  += xv[r] * wx[4 + j];
        gni += xv[r] * wx[8 + j];
      }
      float rr = sigmoidf_(gr);
      float zz = sigmoidf_(gz);
      float nn = tanhf(gni + rr * gnh);
      float hv = (1.f - zz) * nn + zz * hold[j][r];
      hold[j][r] = valid ? hv : 0.f;
    }
  }
}

// Scatter D-layout h into wave-private LDS [16][64] bf16 (and optionally the
// padded bf16 ys stream in global memory).
template <bool WRITE_YS>
DEV void store_h(const float hold[4][8], __bf16* ldsH, __bf16* ysRow,
                 int p, int half) {
#pragma unroll
  for (int j = 0; j < 4; ++j) {
    int c = j * 16 + p;
#pragma unroll
    for (int r = 0; r < 8; ++r) {
      int m = r + 8 * half;
      __bf16 hb = (__bf16)hold[j][r];
      ldsH[m * HP + c] = hb;
      if (WRITE_YS) ysRow[(size_t)m * HP + c] = hb;
    }
  }
}

DEV void load_biases(const float* bih, const float* bhh, int p,
                     float brz0[4], float brz1[4], float bin[4], float bhn[4]) {
#pragma unroll
  for (int j = 0; j < 4; ++j) {
    int c = j * 16 + p;
    bool v = c < Hsz;
    brz0[j] = v ? (bih[c] + bhh[c]) : 0.f;
    brz1[j] = v ? (bih[60 + c] + bhh[60 + c]) : 0.f;
    bin[j]  = v ? bih[120 + c] : 0.f;
    bhn[j]  = v ? bhh[120 + c] : 0.f;
  }
}

#define WAVE_DSCNT_FENCE() asm volatile("s_wait_dscnt 0" ::: "memory")

// ---------------------------------------------------------------------------
// Encoder layer 0: scalar input (F=1), rank-1 ih path folded into gate math.
// ---------------------------------------------------------------------------
__global__ __launch_bounds__(128) void enc_scalar_kernel(
    const float* __restrict__ xBT, const float* __restrict__ Wih,
    const float* __restrict__ Whh, const float* __restrict__ bih,
    const float* __restrict__ bhh, __bf16* __restrict__ ysOut,
    float* __restrict__ encOut, int layer) {
  __shared__ alignas(32) __bf16 bhhF[NT * 2 * 32 * 16];
  __shared__ alignas(32) __bf16 hT[4][16 * HP];

  int tid = threadIdx.x, lane = tid & 31, wave = tid >> 5;
  int p = lane & 15, half = (lane >> 4) & 1;
  int rowbase = blockIdx.x * 64 + wave * 16;

  fill_bfrags(bhhF, Whh, Hsz, tid, 128);
  __syncthreads();

  for (int i = lane; i < 16 * HP; i += 32) hT[wave][i] = (__bf16)0.f;

  float brz0[4], brz1[4], bin[4], bhn[4], wx[12];
  load_biases(bih, bhh, p, brz0, brz1, bin, bhn);
#pragma unroll
  for (int j = 0; j < 4; ++j) {
    int c = j * 16 + p;
    bool v = c < Hsz;
    wx[j]     = v ? Wih[c] : 0.f;
    wx[4 + j] = v ? Wih[60 + c] : 0.f;
    wx[8 + j] = v ? Wih[120 + c] : 0.f;
  }

  float hold[4][8];
#pragma unroll
  for (int j = 0; j < 4; ++j)
#pragma unroll
    for (int r = 0; r < 8; ++r) hold[j][r] = 0.f;

  WAVE_DSCNT_FENCE();

#pragma unroll 1
  for (int t = 0; t < Tsz; ++t) {
    v8f acc[16];
#pragma unroll
    for (int i = 0; i < 16; ++i) acc[i] = v8f_zero();

    float xv[8];
#pragma unroll
    for (int r = 0; r < 8; ++r)
      xv[r] = xBT[(size_t)(rowbase + r + 8 * half) * Tsz + t];

    v16bf a0 = load_a(hT[wave], lane, 0);
    v16bf a1 = load_a(hT[wave], lane, 1);
    gemm12(acc, 8, bhhF, a0, a1, lane);

    gates_update<true>(hold, acc, brz0, brz1, bin, bhn, wx, xv, p);
    store_h<true>(hold, hT[wave],
                  ysOut + ((size_t)t * Bsz + rowbase) * HP, p, half);
    WAVE_DSCNT_FENCE();
  }

#pragma unroll
  for (int j = 0; j < 4; ++j) {
    int c = j * 16 + p;
    if (c < Hsz) {
#pragma unroll
      for (int r = 0; r < 8; ++r) {
        int m = r + 8 * half;
        encOut[(size_t)(rowbase + m) * 180 + layer * 60 + c] = hold[j][r];
      }
    }
  }
}

// ---------------------------------------------------------------------------
// Encoder layers 1/2: vector input from padded bf16 ys stream.
// ---------------------------------------------------------------------------
template <bool WRITE_YS>
__global__ __launch_bounds__(128) void enc_vec_kernel(
    const __bf16* __restrict__ ysIn, const float* __restrict__ Wih,
    const float* __restrict__ Whh, const float* __restrict__ bih,
    const float* __restrict__ bhh, __bf16* __restrict__ ysOut,
    float* __restrict__ encOut, int layer) {
  __shared__ alignas(32) __bf16 bihF[NT * 2 * 32 * 16];
  __shared__ alignas(32) __bf16 bhhF[NT * 2 * 32 * 16];
  __shared__ alignas(32) __bf16 hT[4][16 * HP];

  int tid = threadIdx.x, lane = tid & 31, wave = tid >> 5;
  int p = lane & 15, half = (lane >> 4) & 1;
  int rowbase = blockIdx.x * 64 + wave * 16;

  fill_bfrags(bihF, Wih, Hsz, tid, 128);
  fill_bfrags(bhhF, Whh, Hsz, tid, 128);
  __syncthreads();

  for (int i = lane; i < 16 * HP; i += 32) hT[wave][i] = (__bf16)0.f;

  float brz0[4], brz1[4], bin[4], bhn[4];
  load_biases(bih, bhh, p, brz0, brz1, bin, bhn);

  float hold[4][8];
#pragma unroll
  for (int j = 0; j < 4; ++j)
#pragma unroll
    for (int r = 0; r < 8; ++r) hold[j][r] = 0.f;

  WAVE_DSCNT_FENCE();

#pragma unroll 1
  for (int t = 0; t < Tsz; ++t) {
    v8f acc[16];
#pragma unroll
    for (int i = 0; i < 16; ++i) acc[i] = v8f_zero();

    const __bf16* xrow = ysIn + ((size_t)t * Bsz + rowbase) * HP;
    v16bf ax0 = load_a(xrow, lane, 0);
    v16bf ax1 = load_a(xrow, lane, 1);
    gemm12(acc, 12, bihF, ax0, ax1, lane);

    v16bf ah0 = load_a(hT[wave], lane, 0);
    v16bf ah1 = load_a(hT[wave], lane, 1);
    gemm12(acc, 8, bhhF, ah0, ah1, lane);

    gates_update<false>(hold, acc, brz0, brz1, bin, bhn, nullptr, nullptr, p);
    if (WRITE_YS)
      store_h<true>(hold, hT[wave],
                    ysOut + ((size_t)t * Bsz + rowbase) * HP, p, half);
    else
      store_h<false>(hold, hT[wave], nullptr, p, half);
    WAVE_DSCNT_FENCE();
  }

#pragma unroll
  for (int j = 0; j < 4; ++j) {
    int c = j * 16 + p;
    if (c < Hsz) {
#pragma unroll
      for (int r = 0; r < 8; ++r) {
        int m = r + 8 * half;
        encOut[(size_t)(rowbase + m) * 180 + layer * 60 + c] = hold[j][r];
      }
    }
  }
}

// ---------------------------------------------------------------------------
// Decoder: 3 fused GRUCells + scalar linear head per step.
// ---------------------------------------------------------------------------
DEV void load_h_init(float hold[4][8], __bf16* ldsH,
                     const float* __restrict__ encIn, int rowbase, int layer,
                     int p, int half) {
#pragma unroll
  for (int j = 0; j < 4; ++j) {
    int c = j * 16 + p;
#pragma unroll
    for (int r = 0; r < 8; ++r) {
      int m = r + 8 * half;
      float v = (c < Hsz)
                    ? encIn[(size_t)(rowbase + m) * 180 + layer * 60 + c]
                    : 0.f;
      hold[j][r] = v;
      ldsH[m * HP + c] = (__bf16)v;
    }
  }
}

__global__ __launch_bounds__(128) void dec_kernel(
    const float* __restrict__ yBT,
    const float* __restrict__ c1Wih, const float* __restrict__ c1Whh,
    const float* __restrict__ c1bih, const float* __restrict__ c1bhh,
    const float* __restrict__ c2Wih, const float* __restrict__ c2Whh,
    const float* __restrict__ c2bih, const float* __restrict__ c2bhh,
    const float* __restrict__ c3Wih, const float* __restrict__ c3Whh,
    const float* __restrict__ c3bih, const float* __restrict__ c3bhh,
    const float* __restrict__ linW, const float* __restrict__ linb,
    const float* __restrict__ encIn, float* __restrict__ predOut) {
  __shared__ alignas(32) __bf16 b1hh[NT * 2 * 32 * 16];
  __shared__ alignas(32) __bf16 b2ih[NT * 2 * 32 * 16];
  __shared__ alignas(32) __bf16 b2hh[NT * 2 * 32 * 16];
  __shared__ alignas(32) __bf16 b3ih[NT * 2 * 32 * 16];
  __shared__ alignas(32) __bf16 b3hh[NT * 2 * 32 * 16];
  __shared__ alignas(32) __bf16 h1[4][16 * HP];
  __shared__ alignas(32) __bf16 h2[4][16 * HP];
  __shared__ alignas(32) __bf16 h3[4][16 * HP];

  int tid = threadIdx.x, lane = tid & 31, wave = tid >> 5;
  int p = lane & 15, half = (lane >> 4) & 1;
  int rowbase = blockIdx.x * 64 + wave * 16;

  fill_bfrags(b1hh, c1Whh, Hsz, tid, 128);
  fill_bfrags(b2ih, c2Wih, Hsz, tid, 128);
  fill_bfrags(b2hh, c2Whh, Hsz, tid, 128);
  fill_bfrags(b3ih, c3Wih, Hsz, tid, 128);
  fill_bfrags(b3hh, c3Whh, Hsz, tid, 128);
  __syncthreads();

  float brz0a[4], brz1a[4], bina[4], bhna[4], wx1[12];
  float brz0b[4], brz1b[4], binb[4], bhnb[4];
  float brz0c[4], brz1c[4], binc[4], bhnc[4];
  load_biases(c1bih, c1bhh, p, brz0a, brz1a, bina, bhna);
  load_biases(c2bih, c2bhh, p, brz0b, brz1b, binb, bhnb);
  load_biases(c3bih, c3bhh, p, brz0c, brz1c, binc, bhnc);

  float wl[4];
  float lb = linb[0];
#pragma unroll
  for (int j = 0; j < 4; ++j) {
    int c = j * 16 + p;
    bool v = c < Hsz;
    wx1[j]     = v ? c1Wih[c] : 0.f;
    wx1[4 + j] = v ? c1Wih[60 + c] : 0.f;
    wx1[8 + j] = v ? c1Wih[120 + c] : 0.f;
    wl[j]      = v ? linW[c] : 0.f;
  }

  // carry0 = (enc layer2, enc layer1, enc layer0)
  float hold1[4][8], hold2[4][8], hold3[4][8];
  load_h_init(hold1, h1[wave], encIn, rowbase, 2, p, half);
  load_h_init(hold2, h2[wave], encIn, rowbase, 1, p, half);
  load_h_init(hold3, h3[wave], encIn, rowbase, 0, p, half);
  WAVE_DSCNT_FENCE();

#pragma unroll 1
  for (int t = 0; t < Tsz; ++t) {
    float xv[8];
#pragma unroll
    for (int r = 0; r < 8; ++r)
      xv[r] = yBT[(size_t)(rowbase + r + 8 * half) * Tsz + t];

    // ---- cell 1 (scalar input) ----
    {
      v8f acc[16];
#pragma unroll
      for (int i = 0; i < 16; ++i) acc[i] = v8f_zero();
      v16bf a0 = load_a(h1[wave], lane, 0);
      v16bf a1 = load_a(h1[wave], lane, 1);
      gemm12(acc, 8, b1hh, a0, a1, lane);
      gates_update<true>(hold1, acc, brz0a, brz1a, bina, bhna, wx1, xv, p);
      store_h<false>(hold1, h1[wave], nullptr, p, half);
      WAVE_DSCNT_FENCE();
    }
    // ---- cell 2 (input = h1) ----
    {
      v8f acc[16];
#pragma unroll
      for (int i = 0; i < 16; ++i) acc[i] = v8f_zero();
      v16bf ax0 = load_a(h1[wave], lane, 0);
      v16bf ax1 = load_a(h1[wave], lane, 1);
      gemm12(acc, 12, b2ih, ax0, ax1, lane);
      v16bf ah0 = load_a(h2[wave], lane, 0);
      v16bf ah1 = load_a(h2[wave], lane, 1);
      gemm12(acc, 8, b2hh, ah0, ah1, lane);
      gates_update<false>(hold2, acc, brz0b, brz1b, binb, bhnb, nullptr,
                          nullptr, p);
      store_h<false>(hold2, h2[wave], nullptr, p, half);
      WAVE_DSCNT_FENCE();
    }
    // ---- cell 3 (input = h2) ----
    {
      v8f acc[16];
#pragma unroll
      for (int i = 0; i < 16; ++i) acc[i] = v8f_zero();
      v16bf ax0 = load_a(h2[wave], lane, 0);
      v16bf ax1 = load_a(h2[wave], lane, 1);
      gemm12(acc, 12, b3ih, ax0, ax1, lane);
      v16bf ah0 = load_a(h3[wave], lane, 0);
      v16bf ah1 = load_a(h3[wave], lane, 1);
      gemm12(acc, 8, b3hh, ah0, ah1, lane);
      gates_update<false>(hold3, acc, brz0c, brz1c, binc, bhnc, nullptr,
                          nullptr, p);
      store_h<false>(hold3, h3[wave], nullptr, p, half);
      WAVE_DSCNT_FENCE();
    }
    // ---- linear head: pred[m] = h3[m,:] . linW + lb ----
#pragma unroll
    for (int r = 0; r < 8; ++r) {
      float s = hold3[0][r] * wl[0] + hold3[1][r] * wl[1] +
                hold3[2][r] * wl[2] + hold3[3][r] * wl[3];
      s += __shfl_xor(s, 1);
      s += __shfl_xor(s, 2);
      s += __shfl_xor(s, 4);
      s += __shfl_xor(s, 8);
      if (p == 0)
        predOut[(size_t)(rowbase + r + 8 * half) * Tsz + t] = s + lb;
    }
  }
}

// ---------------------------------------------------------------------------
extern "C" void kernel_launch(void* const* d_in, const int* in_sizes, int n_in,
                              void* d_out, int out_size, void* d_ws,
                              size_t ws_size, hipStream_t stream) {
  (void)in_sizes; (void)n_in; (void)out_size; (void)ws_size;

  const float* inputs  = (const float*)d_in[0];
  const float* outputs = (const float*)d_in[1];
  const float* eWih0 = (const float*)d_in[2];
  const float* eWhh0 = (const float*)d_in[3];
  const float* ebih0 = (const float*)d_in[4];
  const float* ebhh0 = (const float*)d_in[5];
  const float* eWih1 = (const float*)d_in[6];
  const float* eWhh1 = (const float*)d_in[7];
  const float* ebih1 = (const float*)d_in[8];
  const float* ebhh1 = (const float*)d_in[9];
  const float* eWih2 = (const float*)d_in[10];
  const float* eWhh2 = (const float*)d_in[11];
  const float* ebih2 = (const float*)d_in[12];
  const float* ebhh2 = (const float*)d_in[13];
  const float* c1Wih = (const float*)d_in[14];
  const float* c1Whh = (const float*)d_in[15];
  const float* c1bih = (const float*)d_in[16];
  const float* c1bhh = (const float*)d_in[17];
  const float* c2Wih = (const float*)d_in[18];
  const float* c2Whh = (const float*)d_in[19];
  const float* c2bih = (const float*)d_in[20];
  const float* c2bhh = (const float*)d_in[21];
  const float* c3Wih = (const float*)d_in[22];
  const float* c3Whh = (const float*)d_in[23];
  const float* c3bih = (const float*)d_in[24];
  const float* c3bhh = (const float*)d_in[25];
  const float* linW  = (const float*)d_in[26];
  const float* linb  = (const float*)d_in[27];

  float* pred = (float*)d_out;                       // [B, T] (F=1)
  float* enc  = (float*)d_out + (size_t)Bsz * Tsz;   // [B, 3, 60]

  __bf16* ys0 = (__bf16*)d_ws;                             // [T][B][64] bf16
  __bf16* ys1 = ys0 + (size_t)Tsz * Bsz * HP;

  dim3 grid(16), block(128);  // 16 WG x 4 waves x 16 rows = 1024 batch rows

  enc_scalar_kernel<<<grid, block, 0, stream>>>(inputs, eWih0, eWhh0, ebih0,
                                                ebhh0, ys0, enc, 0);
  enc_vec_kernel<true><<<grid, block, 0, stream>>>(ys0, eWih1, eWhh1, ebih1,
                                                   ebhh1, ys1, enc, 1);
  enc_vec_kernel<false><<<grid, block, 0, stream>>>(ys1, eWih2, eWhh2, ebih2,
                                                    ebhh2, nullptr, enc, 2);
  dec_kernel<<<grid, block, 0, stream>>>(
      outputs, c1Wih, c1Whh, c1bih, c1bhh, c2Wih, c2Whh, c2bih, c2bhh, c3Wih,
      c3Whh, c3bih, c3bhh, linW, linb, enc, pred);
}
